// DistanceNeighborSampler_13666585936662
// MI455X (gfx1250) — matrix operand order repaired and checked by reading, not compiled, not run on hardware
//
#include <hip/hip_runtime.h>

// ---------- types ----------
typedef __attribute__((ext_vector_type(16))) __bf16 v16bf;
typedef __attribute__((ext_vector_type(8)))  float  v8f;
typedef __attribute__((ext_vector_type(4)))  float  v4f;

#define FEAT     128
#define NEI      64
#define NS       16
#define TILE     16     // batch rows per workgroup
#define THREADS  256
#define NWAVES   8
#define CB_STRIDE 144   // bf16 elements per row: 288B, multiple of 32B for v16bf loads

// ---------- wave32 cross-lane helpers via ds_swizzle (no LDS memory touched) ----------
template <int PAT>
__device__ __forceinline__ float swzf(float v) {
  return __int_as_float(__builtin_amdgcn_ds_swizzle(__float_as_int(v), PAT));
}
template <int PAT>
__device__ __forceinline__ int swzi(int v) {
  return __builtin_amdgcn_ds_swizzle(v, PAT);
}

// inclusive scan over 32 lanes (rotate shuffles, guarded)
__device__ __forceinline__ float incl_scan32(float x, int lane) {
  float t;
  t = swzf<0xC420>(x); x += (lane >= 1)  ? t : 0.f;   // rot by 1
  t = swzf<0xC440>(x); x += (lane >= 2)  ? t : 0.f;   // rot by 2
  t = swzf<0xC480>(x); x += (lane >= 4)  ? t : 0.f;   // rot by 4
  t = swzf<0xC500>(x); x += (lane >= 8)  ? t : 0.f;   // rot by 8
  t = swzf<0xC600>(x); x += (lane >= 16) ? t : 0.f;   // rot by 16
  return x;
}
// butterfly all-reduce add over 32 lanes (xor masks, and_mask=31)
__device__ __forceinline__ int reduce_add_i32(int v) {
  v += swzi<0x041F>(v);
  v += swzi<0x081F>(v);
  v += swzi<0x101F>(v);
  v += swzi<0x201F>(v);
  v += swzi<0x401F>(v);
  return v;
}

__device__ __forceinline__ float sumsq4(v4f v) {
  return v.x * v.x + v.y * v.y + v.z * v.z + v.w * v.w;
}

__device__ __forceinline__ unsigned hash_u32(unsigned x) {
  x ^= 0x9E3779B9u ^ (42u * 0x85EBCA6Bu);   // fold in seed 42
  x ^= x >> 16; x *= 0x7FEB352Du;
  x ^= x >> 15; x *= 0x846CA68Bu;
  x ^= x >> 16;
  return x;
}

__global__ __launch_bounds__(THREADS)
void dist_sampler_kernel(const int* __restrict__ ids,
                         const float* __restrict__ feat,
                         const int* __restrict__ adj,
                         int* __restrict__ out) {
  __shared__ int   sid[TILE];
  __shared__ float cnorm[TILE];
  __shared__ alignas(32) __bf16 cbf[TILE][CB_STRIDE];   // center feats, bf16, row-major
  __shared__ int   adjt[TILE][NEI];
  __shared__ float probs[TILE][NEI];
  __shared__ float dtile[NWAVES][16][17];               // per-wave D-tile spill

  const int tid  = threadIdx.x;
  const int wave = tid >> 5;
  const int lane = tid & 31;

  // ---------------- Stage 0: gather ids, adjacency, center features ----------------
  if (tid < TILE) {
    sid[tid]   = ids[blockIdx.x * TILE + tid];
    cnorm[tid] = 0.f;
  }
  __syncthreads();

  // adjacency tile: 16 x 64 ints
  for (int i = tid; i < TILE * NEI; i += THREADS) {
    int r = i >> 6, c = i & 63;
    adjt[r][c] = adj[(long)sid[r] * NEI + c];
  }

  // center features: 16 threads per row, 8 contiguous floats each
  {
    int r  = tid >> 4;
    int c0 = (tid & 15) * 8;
    const float* src = feat + (long)sid[r] * FEAT + c0;
    float part = 0.f;
#pragma unroll
    for (int i = 0; i < 8; ++i) {
      float f = src[i];
      part += f * f;
      cbf[r][c0 + i] = (__bf16)f;                 // B-operand: column r, K contiguous
    }
    atomicAdd(&cnorm[r], part);                   // ds_add_f32
  }
  __syncthreads();

  // ---------------- Stage 1: WMMA distances, 8 neighbor slots per wave ----------------
  const int hrow = lane & 15;   // batch row within tile (A row M / B col N)
  const int half = lane >> 4;   // which K-subset this lane supplies
  float (*dt)[17] = dtile[wave];

  // B fragments are invariant across all 8 slots this wave handles: hoist to VGPRs.
  v16bf bfrag[4];
#pragma unroll
  for (int c = 0; c < 4; ++c)
    bfrag[c] = *(const v16bf*)&cbf[hrow][c * 32 + half * 16];

  // Hoist the 8 neighbor ids so the first global gather issues early.
  int nid8[8];
#pragma unroll
  for (int kk = 0; kk < 8; ++kk)
    nid8[kk] = adjt[hrow][wave * 8 + kk];

  const float cn = cnorm[hrow];

  for (int kk = 0; kk < 8; ++kk) {
    const int k = wave * 8 + kk;
    const float* nf = feat + (long)nid8[kk] * FEAT;

    v8f  acc = {};
    float nsq = 0.f;

#pragma unroll
    for (int c = 0; c < 4; ++c) {
      const int base = c * 32 + half * 16;
      // A-fragment source: each lane covers K base..base+15 contiguous
      v4f f0 = *(const v4f*)(nf + base);
      v4f f1 = *(const v4f*)(nf + base + 4);
      v4f f2 = *(const v4f*)(nf + base + 8);
      v4f f3 = *(const v4f*)(nf + base + 12);
      nsq += sumsq4(f0) + sumsq4(f1) + sumsq4(f2) + sumsq4(f3);

      v16bf a;
      a[0]  = (__bf16)f0.x; a[1]  = (__bf16)f0.y; a[2]  = (__bf16)f0.z; a[3]  = (__bf16)f0.w;
      a[4]  = (__bf16)f1.x; a[5]  = (__bf16)f1.y; a[6]  = (__bf16)f1.z; a[7]  = (__bf16)f1.w;
      a[8]  = (__bf16)f2.x; a[9]  = (__bf16)f2.y; a[10] = (__bf16)f2.z; a[11] = (__bf16)f2.w;
      a[12] = (__bf16)f3.x; a[13] = (__bf16)f3.y; a[14] = (__bf16)f3.z; a[15] = (__bf16)f3.w;

      acc = __builtin_amdgcn_wmma_f32_16x16x32_bf16(
          /*neg_a=*/false, a, /*neg_b=*/false, bfrag[c],
          /*c_mod=*/(short)0, acc, /*reuse_a=*/false, /*reuse_b=*/false);
    }

    // combine the two K-half partial neighbor norms (lanes l <-> l+16 hold same row)
    float nsq2 = nsq + swzf<0x401F>(nsq);

    // spill D tile and read the diagonal: D[j][j] = dot(neigh(b_j,k), center_j)
#pragma unroll
    for (int r = 0; r < 8; ++r)
      dt[r + half * 8][hrow] = acc[r];
    __builtin_amdgcn_wave_barrier();
    // Plain LDS read: stays a ds_load_b32 (addrspace(3)); same-wave DS ordering +
    // the wave barrier above guarantee the spilled tile is visible.
    float d2 = dt[hrow][hrow];
    __builtin_amdgcn_wave_barrier();

    if (half == 0) {
      float dist2 = cn + nsq2 - 2.f * d2;
      dist2 = dist2 > 0.f ? dist2 : 0.f;
      float p = __expf(-__builtin_sqrtf(dist2));
      probs[hrow][k] = p;
    }
  }
  __syncthreads();

  // ---------------- Stage 2: normalize + inverse-CDF categorical sampling ----------------
  // wave w handles tile rows 2w and 2w+1
  for (int row = wave * 2; row < wave * 2 + 2; ++row) {
    const float p0 = probs[row][lane];
    const float p1 = probs[row][lane + 32];

    const float s0 = incl_scan32(p0, lane);
    const float s1 = incl_scan32(p1, lane);
    const float tot0 = swzf<0x03E0>(s0);   // broadcast lane 31
    const float tot1 = swzf<0x03E0>(s1);
    const float cdf0 = s0;                 // cdf for slots [0..31]
    const float cdf1 = tot0 + s1;          // cdf for slots [32..63]
    const float total = tot0 + tot1;

    const int gb = blockIdx.x * TILE + row;
#pragma unroll
    for (int s = 0; s < NS; ++s) {
      const unsigned h = hash_u32((unsigned)(gb * NS + s));
      const float u = (float)(h >> 8) * (1.0f / 16777216.0f) * total;  // u in [0,total)
      int cnt = (cdf0 < u ? 1 : 0) + (cdf1 < u ? 1 : 0);
      cnt = reduce_add_i32(cnt);           // idx = #{k : cdf_k < u}
      int idx = cnt > 63 ? 63 : cnt;
      if (lane == 0)
        out[(long)gb * NS + s] = adjt[row][idx];
    }
  }
}

extern "C" void kernel_launch(void* const* d_in, const int* in_sizes, int n_in,
                              void* d_out, int out_size, void* d_ws, size_t ws_size,
                              hipStream_t stream) {
  const int*   ids  = (const int*)d_in[0];
  const float* feat = (const float*)d_in[2];
  const int*   adj  = (const int*)d_in[4];
  int* out = (int*)d_out;

  const int B = in_sizes[0];        // 16384
  const int blocks = B / TILE;      // 1024 workgroups of 256 threads (8 wave32)
  hipLaunchKernelGGL(dist_sampler_kernel, dim3(blocks), dim3(THREADS), 0, stream,
                     ids, feat, adj, out);
}
